// Dcls1d_91156385890465
// MI455X (gfx1250) — matrix operand order repaired and checked by reading, not compiled, not run on hardware
//
#include <hip/hip_runtime.h>

typedef __attribute__((ext_vector_type(8)))  _Float16 v8h;
typedef __attribute__((ext_vector_type(16))) _Float16 v16h;
typedef __attribute__((ext_vector_type(8)))  float    v8f;

#define DKS 33
#define PADL 16
#define OC 256
#define ICH 256
#define KC 16
#define NB 8
#define TLEN 2048
#define TP (TLEN + 2*PADL)          /* 2080 padded rows */
#define KERN_ELEMS ((size_t)DKS*OC*ICH)

#define LROW 40                      /* padded LDS row pitch in f16 (80 B) */
#define GD   11                      /* taps per staged A group (33 = 3*11) */
#define NG   3                       /* number of tap groups */

// ---------------------------------------------------------------------------
// Phase 1: build dense f16 kernel  kern[d][o][i]  (faithful ConstructKernel1d,
// including the reference quirk that frac comes from out-channel 0's P).
// ---------------------------------------------------------------------------
__global__ void dcls_build_kern(const float* __restrict__ weight,
                                const float* __restrict__ P,
                                _Float16* __restrict__ kern) {
  const int o = blockIdx.x;      // 0..255
  const int i = threadIdx.x;     // 0..255
  float w[KC], fr[KC];
  int   pf[KC];
  const float* wp = weight + ((size_t)o*ICH + i)*KC;
  const float* pp = P      + ((size_t)o*ICH + i)*KC;
  const float* p0 = P      + (size_t)i*KC;            // out-channel 0 row
#pragma unroll
  for (int k = 0; k < KC; ++k) {
    w[k] = wp[k];
    float p = pp[k] + (float)(DKS/2);
    pf[k] = (int)floorf(p);
    float q = p0[k] + (float)(DKS/2);
    fr[k] = q - floorf(q);
  }
  _Float16* kr = kern + (size_t)o*ICH + i;
  for (int d = 0; d < DKS; ++d) {
    float a = 0.f;
#pragma unroll
    for (int k = 0; k < KC; ++k) {
      if (d == pf[k])     a += w[k] * (1.0f - fr[k]);
      if (d == pf[k] + 1) a += w[k] * fr[k];
    }
    kr[(size_t)d*OC*ICH] = (_Float16)a;
  }
}

// ---------------------------------------------------------------------------
// Phase 2: transpose + pad + f16-convert x:  xT[n][tp][i],  tp = t + 16,
// zero rows for tp<16 or tp>=2064.  LDS 16x16 tile transpose.
// ---------------------------------------------------------------------------
__global__ void dcls_xT(const float* __restrict__ x,
                        _Float16* __restrict__ xT) {
  __shared__ _Float16 tile[16][16 + 8];
  const int n     = blockIdx.z;
  const int itile = blockIdx.x * 16;
  const int ttile = blockIdx.y * 16;          // padded-row base
  const int tid   = threadIdx.x;              // 256 threads
  {
    const int tr = tid & 15, ir = tid >> 4;
    const int tg = ttile + tr - PADL;         // source time index
    const int ig = itile + ir;
    float v = 0.f;
    if (tg >= 0 && tg < TLEN) v = x[((size_t)n*ICH + ig)*TLEN + tg];
    tile[ir][tr] = (_Float16)v;
  }
  __syncthreads();
  {
    const int iw = tid & 15, tw = tid >> 4;
    xT[((size_t)n*TP + ttile + tw)*ICH + itile + iw] = tile[iw][tw];
  }
}

// ---------------------------------------------------------------------------
// Phase 3: implicit-GEMM conv on WMMA with grouped LDS staging.
//   out[n,o,t] = sum_{i,d} kern[d][o][i] * xT[n][t+d][i] + bias[o]
// Block: 256 threads = 8 waves (2 M-waves x 4 N-waves); tile 64(o) x 128(t).
// Per ic-chunk: stage the 160-row x-window once (all 33 taps reuse it);
// stage A in double-buffered groups of 11 taps -> 44 WMMAs per barrier.
// ---------------------------------------------------------------------------
__global__ __launch_bounds__(256)
void dcls_conv_wmma(const _Float16* __restrict__ kern,
                    const _Float16* __restrict__ xT,
                    const float* __restrict__ bias,
                    float* __restrict__ out) {
  __shared__ _Float16 lsA[2][GD][64][LROW]; // double-buffered A tap-group
  __shared__ _Float16 lsX[160][LROW];       // x window: rows tblk..tblk+160

  const int tid  = threadIdx.x;
  const int lane = tid & 31;
  const int wave = tid >> 5;
  const int wm   = wave >> 2;        // 0..1  (o half)
  const int wn   = wave & 3;         // 0..3  (t quarter)
  const int half = lane >> 4;
  const int l16  = lane & 15;
  const int n    = blockIdx.z;
  const int oblk = blockIdx.x * 64;
  const int tblk = blockIdx.y * 128;
  const int o0   = oblk + wm * 32;   // + mt*16
  const int t0l  = wn * 32;          // local t base within window, + nt*16

  // cooperative-stage coordinates (one b128 per thread per tap for A)
  const int sA_o = tid >> 2;
  const int sA_i = (tid & 3) * 8;

  const _Float16* xrow_base = xT + (size_t)n * TP * ICH;
  const _Float16* kern_st   = kern + ((size_t)(oblk + sA_o))*ICH + sA_i;

  v8f c[2][2] = {};

  for (int ic = 0; ic < ICH; ic += 32) {
    // ---- stage X window: 160 rows x 32 ch (32 B per idx-slot) ----
    for (int idx = tid; idx < 320; idx += 256) {
      const int row = idx >> 1;
      const int io  = (idx & 1) * 16;
      const _Float16* src = xrow_base + ((size_t)(tblk + row))*ICH + ic + io;
      v8h lo = *(const v8h*)src;
      v8h hi = *(const v8h*)(src + 8);
      *(v8h*)&lsX[row][io]     = lo;
      *(v8h*)&lsX[row][io + 8] = hi;
    }
    // ---- stage A group 0 (taps 0..GD-1) into buffer 0 ----
#pragma unroll
    for (int j = 0; j < GD; ++j) {
      v8h v = *(const v8h*)(kern_st + (size_t)j*OC*ICH + ic);
      *(v8h*)&lsA[0][j][sA_o][sA_i] = v;
    }
    __syncthreads();

    for (int g = 0; g < NG; ++g) {
      const int cur = g & 1;
      const bool pf = (g + 1 < NG);
#pragma unroll
      for (int j = 0; j < GD; ++j) {
        const int d = g*GD + j;
        // prefetch next group's tap j (overlaps this step's WMMAs)
        v8h pre = {};
        if (pf)
          pre = *(const v8h*)(kern_st + (size_t)(d + GD)*OC*ICH + ic);

        // ---- A fragments from LDS (ISA 7.12.2 16-bit A layout) ----
        v16h a[2];
#pragma unroll
        for (int mt = 0; mt < 2; ++mt) {
          const _Float16* ap = &lsA[cur][j][wm*32 + mt*16 + l16][0];
          ((v8h*)&a[mt])[0] = *(const v8h*)(ap + half*8);      // K=half*8+0..7
          ((v8h*)&a[mt])[1] = *(const v8h*)(ap + 16 + half*8); // K=16+half*8+..
        }
        // ---- B fragments from LDS x-window, row shifted by tap d ----
        v16h b[2];
#pragma unroll
        for (int nt = 0; nt < 2; ++nt) {
          const _Float16* bp = &lsX[t0l + nt*16 + l16 + d][0];
          ((v8h*)&b[nt])[0] = *(const v8h*)(bp + half*8);      // K=half*8+0..7
          ((v8h*)&b[nt])[1] = *(const v8h*)(bp + 16 + half*8); // K=16+half*8+..
        }
#pragma unroll
        for (int mt = 0; mt < 2; ++mt)
#pragma unroll
          for (int nt = 0; nt < 2; ++nt)
            c[mt][nt] = __builtin_amdgcn_wmma_f32_16x16x32_f16(
                false, a[mt], false, b[nt], (short)0, c[mt][nt], false, false);

        // ---- commit prefetched tap to the other buffer ----
        if (pf)
          *(v8h*)&lsA[1 - cur][j][sA_o][sA_i] = pre;
      }
      __syncthreads();   // group boundary: stores visible, reads drained
    }
  }

  // ---- epilogue: C/D layout -> o = o0+mt*16+half*8+r, t = tblk+... ----
#pragma unroll
  for (int mt = 0; mt < 2; ++mt) {
#pragma unroll
    for (int nt = 0; nt < 2; ++nt) {
      const int t = tblk + t0l + nt*16 + l16;
#pragma unroll
      for (int r = 0; r < 8; ++r) {
        const int o = o0 + mt*16 + half*8 + r;
        out[((size_t)n*OC + o)*TLEN + t] = c[mt][nt][r] + bias[o];
      }
    }
  }
}

// ---------------------------------------------------------------------------
extern "C" void kernel_launch(void* const* d_in, const int* in_sizes, int n_in,
                              void* d_out, int out_size, void* d_ws, size_t ws_size,
                              hipStream_t stream) {
  const float* x      = (const float*)d_in[0];   // (8,256,2048)
  const float* weight = (const float*)d_in[1];   // (256,256,16)
  const float* P      = (const float*)d_in[2];   // (1,256,256,16)
  const float* bias   = (const float*)d_in[3];   // (256,)
  float* out = (float*)d_out;

  _Float16* kern = (_Float16*)d_ws;              // 33*256*256 f16 = 4.3 MB
  _Float16* xTp  = kern + KERN_ELEMS;            // 8*2080*256 f16 = 8.5 MB

  dcls_build_kern<<<dim3(OC), dim3(ICH), 0, stream>>>(weight, P, kern);
  dcls_xT<<<dim3(ICH/16, TP/16, NB), dim3(256), 0, stream>>>(x, xTp);
  dcls_conv_wmma<<<dim3(OC/64, TLEN/128, NB), dim3(256), 0, stream>>>(kern, xTp, bias, out);
}
